// SelectiveOutputLM_57681410785815
// MI455X (gfx1250) — matrix operand order repaired and checked by reading, not compiled, not run on hardware
//
#include <hip/hip_runtime.h>
#include <hip/hip_bf16.h>
#include <math.h>

// ---------------------------------------------------------------------------
// SelectiveOutputLM on MI455X (gfx1250): bf16 WMMA pipeline.
//   h = embed_in[ids]
//   stack: 6x { y=LN(h); h += gelu(y@W1+b1)@W2+b2 }
//   h = h@hp_w^T + hp_b ; stack again
//   logits = LN_f(h) @ embed_out_w^T ; final_hidden = h[:, -1, :]
// GEMMs: v_wmma_f32_16x16x32_bf16, 64x64 per wave (4x4 tiles), 128x128/block.
// LayerNorm: row staged to LDS via GLOBAL_LOAD_ASYNC_TO_LDS_B128 if available.
// ---------------------------------------------------------------------------

typedef __attribute__((ext_vector_type(16))) __bf16 v16bf;
typedef __attribute__((ext_vector_type(8)))  float  v8f;
typedef __attribute__((ext_vector_type(4)))  unsigned int u32x4;
typedef __attribute__((ext_vector_type(4)))  int v4i;

static constexpr int Vv = 50304, Hh = 512, Ll = 6, Ff = 2048;
static constexpr int Bb = 4, Ss = 1024, Mm = Bb * Ss;   // 4096 tokens

#if defined(__gfx1250__) && __has_builtin(__builtin_amdgcn_global_load_async_to_lds_b128)
#define HAVE_ASYNC_LDS 1
#else
#define HAVE_ASYNC_LDS 0
#endif

#if HAVE_ASYNC_LDS
typedef __attribute__((address_space(1))) v4i GV4;   // global <4 x i32>
typedef __attribute__((address_space(3))) v4i LV4;   // LDS    <4 x i32>
#endif

// ---- helpers --------------------------------------------------------------

__device__ __forceinline__ unsigned short f2bf(float f) {
    unsigned int u = __float_as_uint(f);
    u += 0x7FFFu + ((u >> 16) & 1u);          // round-to-nearest-even
    return (unsigned short)(u >> 16);
}

__device__ __forceinline__ float gelu_tanh(float x) {
    const float c0 = 0.7978845608028654f;     // sqrt(2/pi)
    const float c1 = 0.044715f;
    float x3 = x * x * x;
    return 0.5f * x * (1.0f + tanhf(c0 * (x + c1 * x3)));
}

union FragU { struct { u32x4 lo, hi; } q; v16bf v; };

// A fragment (16x32 MxK, bf16): lane r=L&15, half=L>>4.
// V0-3 <- K = half*8 + 0..7 ; V4-7 <- K = 16 + half*8 + 0..7 (ISA 7.12.2)
__device__ __forceinline__ v16bf load_a_frag(const unsigned short* __restrict__ base,
                                             int row0, int k0, int ldk, int lane) {
    int r = lane & 15, half = lane >> 4;
    const unsigned short* p = base + (size_t)(row0 + r) * ldk + k0 + half * 8;
    FragU f;
    f.q.lo = *(const u32x4*)p;
    f.q.hi = *(const u32x4*)(p + 16);
    return f.v;
}

// B fragment (32x16 KxN, bf16) from Bt stored N x K row-major:
// lane n=L&15: V0-3 <- K = half*8 + 0..7 ; V4-7 <- K = 16 + half*8 + 0..7 (matches A).
__device__ __forceinline__ v16bf load_b_frag(const unsigned short* __restrict__ base,
                                             int col0, int k0, int ldk, int lane) {
    int n = lane & 15, half = lane >> 4;
    const unsigned short* p = base + (size_t)(col0 + n) * ldk + k0 + half * 8;
    FragU f;
    f.q.lo = *(const u32x4*)p;
    f.q.hi = *(const u32x4*)(p + 16);
    return f.v;
}

// ---- GEMM: out(MxN) = A(MxK,bf16) x Bt(NxK,bf16)^T  -----------------------
// Block: 128 threads = 4 waves (2x2), 128x128 block tile, 64x64 per wave.
// EPI 0: out_bf16 = bf16(gelu(acc + bias))
// EPI 1: outF    += acc + bias            (residual, in-place on h)
// EPI 2: outF     = acc + bias            (bias may be null)
template <int EPI>
__global__ void __launch_bounds__(128)
gemm_bf16(const unsigned short* __restrict__ A,
          const unsigned short* __restrict__ Bt,
          const float* __restrict__ bias,
          float* __restrict__ outF,
          unsigned short* __restrict__ outB,
          int M, int N, int K)
{
    const int lane = threadIdx.x & 31;
    const int wave = threadIdx.x >> 5;
    const int m_base = blockIdx.y * 128 + (wave >> 1) * 64;
    const int n_base = blockIdx.x * 128 + (wave & 1) * 64;

    v8f zero = {};
    v8f c[4][4];
#pragma unroll
    for (int i = 0; i < 4; ++i)
#pragma unroll
        for (int j = 0; j < 4; ++j) c[i][j] = zero;

    for (int k0 = 0; k0 < K; k0 += 32) {
        if (k0 + 32 < K) {
            // next K slab of both streamed operands -> global_prefetch_b8
            __builtin_prefetch(Bt + (size_t)(n_base + (lane & 15)) * K + k0 + 32, 0, 1);
            __builtin_prefetch(A  + (size_t)(m_base + (lane & 15)) * K + k0 + 32, 0, 1);
        }
        v16bf a[4], b[4];
#pragma unroll
        for (int t = 0; t < 4; ++t) a[t] = load_a_frag(A,  m_base + t * 16, k0, K, lane);
#pragma unroll
        for (int t = 0; t < 4; ++t) b[t] = load_b_frag(Bt, n_base + t * 16, k0, K, lane);

#pragma unroll
        for (int i = 0; i < 4; ++i)
#pragma unroll
            for (int j = 0; j < 4; ++j)
                c[i][j] = __builtin_amdgcn_wmma_f32_16x16x32_bf16(
                    false, a[i], false, b[j], (short)0, c[i][j], false, false);
    }

    // C/D layout: lane L -> N = n_tile + (L&15); VGPR r -> M = m_tile + r + 8*(L>>4)
#pragma unroll
    for (int i = 0; i < 4; ++i) {
#pragma unroll
        for (int j = 0; j < 4; ++j) {
            int n  = n_base + j * 16 + (lane & 15);
            float bv = bias ? bias[n] : 0.0f;
            int m0 = m_base + i * 16 + (lane >> 4) * 8;
#pragma unroll
            for (int r = 0; r < 8; ++r) {
                size_t idx = (size_t)(m0 + r) * N + n;
                float v = c[i][j][r] + bv;
                if (EPI == 0)      outB[idx]  = f2bf(gelu_tanh(v));
                else if (EPI == 1) outF[idx] += v;
                else               outF[idx]  = v;
            }
        }
    }
}

// ---- LayerNorm (block per token row), emits bf16 for next GEMM ------------
// Row (512 f32 = 2KB) is staged into LDS once (async copy when available),
// then read three times from LDS instead of three global passes.
__global__ void __launch_bounds__(128)
layernorm_to_bf16(const float* __restrict__ h, const float* __restrict__ w,
                  const float* __restrict__ b, unsigned short* __restrict__ out)
{
    __shared__ float xs[Hh];
    __shared__ float red[128];
    const int row = blockIdx.x;
    const int tid = threadIdx.x;
    const float* x = h + (size_t)row * Hh;

#if HAVE_ASYNC_LDS
    {
        // each lane copies 16B: 128 lanes x 16B = whole 2KB row
        GV4* gsrc = (GV4*)(x + tid * 4);
        LV4* ldst = (LV4*)(&xs[tid * 4]);
        __builtin_amdgcn_global_load_async_to_lds_b128(gsrc, ldst, 0, 0);
#if __has_builtin(__builtin_amdgcn_s_wait_asynccnt)
        __builtin_amdgcn_s_wait_asynccnt(0);
#else
        asm volatile("s_wait_asynccnt 0x0" ::: "memory");
#endif
    }
#else
    {
        const float4* src = (const float4*)x;
        ((float4*)xs)[tid] = src[tid];
    }
#endif
    __syncthreads();

    float s = 0.f;
    for (int i = tid; i < Hh; i += 128) s += xs[i];
    red[tid] = s; __syncthreads();
    for (int off = 64; off > 0; off >>= 1) { if (tid < off) red[tid] += red[tid + off]; __syncthreads(); }
    float mu = red[0] / (float)Hh;
    __syncthreads();

    float vs = 0.f;
    for (int i = tid; i < Hh; i += 128) { float d = xs[i] - mu; vs += d * d; }
    red[tid] = vs; __syncthreads();
    for (int off = 64; off > 0; off >>= 1) { if (tid < off) red[tid] += red[tid + off]; __syncthreads(); }
    float inv = rsqrtf(red[0] / (float)Hh + 1e-5f);

    for (int i = tid; i < Hh; i += 128)
        out[(size_t)row * Hh + i] = f2bf((xs[i] - mu) * inv * w[i] + b[i]);
}

// ---- misc data-movement kernels -------------------------------------------
__global__ void embed_gather(const int* __restrict__ ids, const float* __restrict__ emb,
                             float* __restrict__ h)
{
    int row = blockIdx.x;                     // token index 0..Mm-1
    int id  = ids[row];
    for (int i = threadIdx.x; i < Hh; i += blockDim.x)
        h[(size_t)row * Hh + i] = emb[(size_t)id * Hh + i];
}

__global__ void conv_f32_bf16(const float* __restrict__ in, unsigned short* __restrict__ out,
                              size_t n)
{
    size_t i = (size_t)blockIdx.x * blockDim.x + threadIdx.x;
    if (i < n) out[i] = f2bf(in[i]);
}

// in: (L, R, C) fp32  ->  out: (L, C, R) bf16   (indexed from output)
__global__ void transpose_bf16(const float* __restrict__ in, unsigned short* __restrict__ out,
                               int R, int C, int Lcount)
{
    size_t i = (size_t)blockIdx.x * blockDim.x + threadIdx.x;
    size_t tot = (size_t)Lcount * R * C;
    if (i >= tot) return;
    size_t rc  = (size_t)R * C;
    int l   = (int)(i / rc);
    int rem = (int)(i % rc);
    int c   = rem / R;
    int r   = rem % R;
    out[i] = f2bf(in[(size_t)l * rc + (size_t)r * C + c]);
}

__global__ void copy_final_hidden(const float* __restrict__ h, float* __restrict__ out)
{
    int i = blockIdx.x * blockDim.x + threadIdx.x;
    if (i < Bb * Hh) {
        int b = i / Hh, c = i % Hh;
        out[i] = h[(size_t)(b * Ss + (Ss - 1)) * Hh + c];
    }
}

// ---------------------------------------------------------------------------
extern "C" void kernel_launch(void* const* d_in, const int* in_sizes, int n_in,
                              void* d_out, int out_size, void* d_ws, size_t ws_size,
                              hipStream_t stream)
{
    (void)in_sizes; (void)n_in; (void)out_size; (void)ws_size;

    const int*   ids   = (const int*)d_in[0];
    const float* emb   = (const float*)d_in[1];
    const float* hp_w  = (const float*)d_in[2];
    const float* hp_b  = (const float*)d_in[3];
    const float* ln_w  = (const float*)d_in[4];
    const float* ln_b  = (const float*)d_in[5];
    const float* W1    = (const float*)d_in[6];
    const float* b1    = (const float*)d_in[7];
    const float* W2    = (const float*)d_in[8];
    const float* b2    = (const float*)d_in[9];
    const float* lnf_w = (const float*)d_in[10];
    const float* lnf_b = (const float*)d_in[11];
    const float* Eo    = (const float*)d_in[12];

    // Workspace layout (256B aligned so 16B fragment loads stay aligned)
    char* ws = (char*)d_ws;
    size_t off = 0;
    auto alloc = [&](size_t bytes) {
        void* p = ws + off;
        off = (off + bytes + 255) & ~(size_t)255;
        return p;
    };
    unsigned short* W1t = (unsigned short*)alloc((size_t)Ll * Hh * Ff * 2); // (L,F,H)
    unsigned short* W2t = (unsigned short*)alloc((size_t)Ll * Hh * Ff * 2); // (L,H,F)
    unsigned short* HPt = (unsigned short*)alloc((size_t)Hh * Hh * 2);      // hp_w as-is (N x K)
    unsigned short* EOt = (unsigned short*)alloc((size_t)Vv * Hh * 2);      // embed_out as-is (N x K)
    float*          h   = (float*)         alloc((size_t)Mm * Hh * 4);
    unsigned short* ybf = (unsigned short*)alloc((size_t)Mm * Hh * 2);
    unsigned short* tbf = (unsigned short*)alloc((size_t)Mm * Ff * 2);

    // --- one-time per call weight conversion -------------------------------
    {
        size_t n = (size_t)Ll * Hh * Ff;
        unsigned g = (unsigned)((n + 255) / 256);
        transpose_bf16<<<g, 256, 0, stream>>>(W1, W1t, Hh, Ff, Ll);   // (L,H,F)->(L,F,H)
        transpose_bf16<<<g, 256, 0, stream>>>(W2, W2t, Ff, Hh, Ll);   // (L,F,H)->(L,H,F)
    }
    {
        size_t n = (size_t)Hh * Hh;
        conv_f32_bf16<<<(unsigned)((n + 255) / 256), 256, 0, stream>>>(hp_w, HPt, n);
    }
    {
        size_t n = (size_t)Vv * Hh;
        conv_f32_bf16<<<(unsigned)((n + 255) / 256), 256, 0, stream>>>(Eo, EOt, n);
    }

    // --- forward pass -------------------------------------------------------
    embed_gather<<<Mm, 128, 0, stream>>>(ids, emb, h);

    auto run_stack = [&]() {
        for (int l = 0; l < Ll; ++l) {
            layernorm_to_bf16<<<Mm, 128, 0, stream>>>(h, ln_w + (size_t)l * Hh,
                                                      ln_b + (size_t)l * Hh, ybf);
            // t = bf16(gelu(y @ W1 + b1))   (4096 x 2048, K=512)
            gemm_bf16<0><<<dim3(Ff / 128, Mm / 128), 128, 0, stream>>>(
                ybf, W1t + (size_t)l * Ff * Hh, b1 + (size_t)l * Ff,
                nullptr, tbf, Mm, Ff, Hh);
            // h += t @ W2 + b2              (4096 x 512, K=2048)
            gemm_bf16<1><<<dim3(Hh / 128, Mm / 128), 128, 0, stream>>>(
                tbf, W2t + (size_t)l * Hh * Ff, b2 + (size_t)l * Hh,
                h, nullptr, Mm, Hh, Ff);
        }
    };

    run_stack();

    // h = h @ hp_w^T + hp_b   (4096 x 512, K=512); hp_w already N x K
    {
        size_t n = (size_t)Mm * Hh;
        conv_f32_bf16<<<(unsigned)((n + 255) / 256), 256, 0, stream>>>(h, ybf, n);
        gemm_bf16<2><<<dim3(Hh / 128, Mm / 128), 128, 0, stream>>>(
            ybf, HPt, hp_b, h, nullptr, Mm, Hh, Hh);
    }

    run_stack();

    // logits = LN_f(h) @ embed_out^T   (4096 x 50304, K=512)
    layernorm_to_bf16<<<Mm, 128, 0, stream>>>(h, lnf_w, lnf_b, ybf);
    gemm_bf16<2><<<dim3(Vv / 128, Mm / 128), 128, 0, stream>>>(
        ybf, EOt, nullptr, (float*)d_out, nullptr, Mm, Vv, Hh);

    // final_hidden = h[:, -1, :]
    copy_final_hidden<<<(Bb * Hh + 255) / 256, 256, 0, stream>>>(
        h, (float*)d_out + (size_t)Mm * Vv);
}
